// UserEncoder_44186623541748
// MI455X (gfx1250) — matrix operand order repaired and checked by reading, not compiled, not run on hardware
//
#include <hip/hip_runtime.h>

typedef __attribute__((ext_vector_type(16))) __bf16 v16bf;
typedef __attribute__((ext_vector_type(8)))  float  v8f;

constexpr int S  = 50;    // sequence length
constexpr int D  = 256;   // hidden size
constexpr int H2 = 128;   // att_fc1 out
constexpr int SP = 64;    // S padded to 4 WMMA M-tiles
constexpr int XS = 264;   // x row stride in bf16 elems (528 B: 16B-aligned, bank-skewed)
constexpr int ES = 132;   // e-partial row stride in floats (rows +8 land 32 banks away)

struct Frag32B { uint4 lo, hi; };   // 32 bytes -> v16bf

__device__ __forceinline__ unsigned pk_bf16(float lo, float hi) {
  __bf16 l = (__bf16)lo;                       // native v_cvt, RNE
  __bf16 h = (__bf16)hi;
  unsigned short ul = __builtin_bit_cast(unsigned short, l);
  unsigned short uh = __builtin_bit_cast(unsigned short, h);
  return (unsigned)ul | ((unsigned)uh << 16);
}
__device__ __forceinline__ float bfu2f(unsigned short h) {
  unsigned u = ((unsigned)h) << 16;
  return __builtin_bit_cast(float, u);
}
__device__ __forceinline__ float fast_tanh(float x) {
  // tanh(x) = 1 - 2/(e^{2x}+1); exact limits at +/-inf via exp->{inf,0}
  float e = __expf(2.0f * x);
  return 1.0f - 2.0f * __builtin_amdgcn_rcpf(e + 1.0f);
}

// ---- Prologue: convert W1 fp32->bf16 and pre-swizzle into WMMA B-fragment
// layout: flat[(w*8+k)*32 + lane][j] with j = 16 consecutive K values.
// lane<16: K=32k+0..15, col=16w+lane ; lane>=16: K=32k+16..31, col=16w+lane-16.
__global__ __launch_bounds__(256) void pack_w1_kernel(
    const float* __restrict__ W1, unsigned short* __restrict__ w1p) {
  int i    = blockIdx.x * 256 + threadIdx.x;   // 0..32767
  int j    = i & 15;
  int lane = (i >> 4) & 31;
  int k    = (i >> 9) & 7;
  int w    = (i >> 12) & 7;
  int K    = 32 * k + 16 * (lane >> 4) + j;
  int col  = 16 * w + (lane & 15);
  __bf16 h = (__bf16)W1[K * H2 + col];
  w1p[i] = __builtin_bit_cast(unsigned short, h);
}

__global__ __launch_bounds__(256) void user_encoder_kernel(
    const float* __restrict__ un,            // [B,S,D]
    const int*   __restrict__ lm,            // [B,S]
    const float* __restrict__ pad,           // [1,D]
    const unsigned short* __restrict__ w1p,  // pre-swizzled bf16 W1 fragments
    const float* __restrict__ b1,            // [H2]
    const float* __restrict__ W2,            // [H2,1]
    const float* __restrict__ b2,            // [1]
    float* __restrict__ out)                 // [B,D]
{
  __shared__ alignas(16) unsigned short xlds[SP * XS];  // x[b] as bf16, padded rows zero
  __shared__ alignas(16) float elds[SP * ES];           // tanh(e)*W2 partials [row][h]
  __shared__ float alpha_sh[SP];
  __shared__ float rdenom_sh;

  const int tid = threadIdx.x;
  const int b   = blockIdx.x;

  // ---- Stage x[b] (mask/pad applied) into LDS as packed bf16 quads ----
  // thread owns column quad dq (cols 4dq..4dq+3) for a 16-row group:
  // b128 global loads (512 B/wave/row, coalesced), b64 LDS stores (conflict-free).
  {
    const int dq = tid & 63;
    const int s0 = (tid >> 6) * 16;
    const float4  padv = ((const float4*)pad)[dq];
    const float4* xb   = (const float4*)(un + (size_t)b * S * D);
    const int*    mb   = lm + (size_t)b * S;
    uint2* xrow = (uint2*)xlds;                // row stride 528 B = 66 uint2
    #pragma unroll 4
    for (int s = s0; s < s0 + 16; ++s) {
      uint2 pkd{0u, 0u};
      if (s < S) {
        float4 g = xb[s * (D / 4) + dq];
        int    m = mb[s];
        float v0 = m ? g.x : padv.x;           // exact: mask is 0/1
        float v1 = m ? g.y : padv.y;
        float v2 = m ? g.z : padv.z;
        float v3 = m ? g.w : padv.w;
        pkd.x = pk_bf16(v0, v1);
        pkd.y = pk_bf16(v2, v3);
      }
      xrow[s * (XS / 4) + dq] = pkd;
    }
  }

  const int wave = tid >> 5;
  const int lane = tid & 31;
  const int half = lane >> 4;
  const int ln   = lane & 15;
  const int col  = wave * 16 + ln;       // this lane's H2 column

  // ---- B fragments: one 32B contiguous read per (chunk, lane), coalesced ----
  v16bf Bf[8];
  {
    const uint4* wp = (const uint4*)w1p;
    #pragma unroll
    for (int k = 0; k < 8; ++k) {
      int idx = ((wave * 8 + k) * 32 + lane) * 2;
      Frag32B fb{wp[idx], wp[idx + 1]};
      Bf[k] = __builtin_bit_cast(v16bf, fb);
    }
  }
  const float b1v = b1[col];
  const float w2v = W2[col];
  const float b2v = b2[0];

  __syncthreads();                       // x tile ready

  // A-fragment tile loader: lane<16 row M=ln, K = 32k+0..15; lane>=16: +16
  auto load_tile = [&](v16bf* dst, int mt) {
    const unsigned short* arow = &xlds[(mt * 16 + ln) * XS + half * 16];
    #pragma unroll
    for (int k = 0; k < 8; ++k) {
      Frag32B fa;
      fa.lo = *(const uint4*)(arow + 32 * k);        // K = 32k+16h+0..7
      fa.hi = *(const uint4*)(arow + 32 * k + 8);    // K = 32k+16h+8..15
      dst[k] = __builtin_bit_cast(v16bf, fa);
    }
  };

  // ---- GEMM, double-buffered A tiles; compiler pipelines ds loads vs WMMA ----
  v16bf Af[2][8];
  load_tile(Af[0], 0);
  #pragma unroll
  for (int mt = 0; mt < 4; ++mt) {
    if (mt < 3) load_tile(Af[(mt + 1) & 1], mt + 1);   // prefetch next tile
    v8f acc = {};
    #pragma unroll
    for (int k = 0; k < 8; ++k)
      acc = __builtin_amdgcn_wmma_f32_16x16x32_bf16(
                false, Af[mt & 1][k], false, Bf[k], (short)0, acc, false, false);
    // C tile: row = 16*mt + r + 8*half, col = ln. Independent chains per r;
    // conflict-free scatter (rows +8 are 32 banks apart at stride ES=132).
    #pragma unroll
    for (int r = 0; r < 8; ++r) {
      float t = fast_tanh(acc[r] + b1v);
      elds[(mt * 16 + half * 8 + r) * ES + col] = t * w2v;
    }
  }
  __syncthreads();

  // ---- logits -> alpha = exp(logit + b2); padded rows excluded ----
  if (tid < SP) {
    const float4* rowp = (const float4*)&elds[tid * ES];
    float l = 0.f;
    #pragma unroll
    for (int q = 0; q < H2 / 4; ++q) {
      float4 v = rowp[q];
      l += v.x + v.y + v.z + v.w;
    }
    alpha_sh[tid] = (tid < S) ? __expf(l + b2v) : 0.f;
  }
  __syncthreads();
  if (tid == 0) {
    float sum = 0.f;
    for (int s = 0; s < S; ++s) sum += alpha_sh[s];
    rdenom_sh = 1.0f / (sum + 1e-8f);
  }
  __syncthreads();

  // ---- out[b,d] = sum_s x[s,d] * alpha[s] / denom (fp32 accumulate) ----
  float accf = 0.f;
  #pragma unroll 5
  for (int s = 0; s < S; ++s)
    accf += bfu2f(xlds[s * XS + tid]) * alpha_sh[s];
  out[(size_t)b * D + tid] = accf * rdenom_sh;
}

extern "C" void kernel_launch(void* const* d_in, const int* in_sizes, int n_in,
                              void* d_out, int out_size, void* d_ws, size_t ws_size,
                              hipStream_t stream) {
  (void)n_in; (void)out_size; (void)ws_size;
  const float* un  = (const float*)d_in[0];
  const int*   lm  = (const int*)d_in[1];
  const float* pad = (const float*)d_in[2];
  const float* W1  = (const float*)d_in[3];
  const float* b1  = (const float*)d_in[4];
  const float* W2  = (const float*)d_in[5];
  const float* b2  = (const float*)d_in[6];
  float* out = (float*)d_out;
  unsigned short* w1p = (unsigned short*)d_ws;   // 32768 bf16 = 64 KB

  const int B = in_sizes[1] / S;   // log_mask has B*S elements

  pack_w1_kernel<<<(D * H2) / 256, 256, 0, stream>>>(W1, w1p);
  user_encoder_kernel<<<B, 256, 0, stream>>>(un, lm, pad, w1p, b1, W2, b2, out);
}
